// GraphAttentionEncoder_13151189860790
// MI455X (gfx1250) — matrix operand order repaired and checked by reading, not compiled, not run on hardware
//
#include <hip/hip_runtime.h>
#include <hip/hip_bf16.h>

#define BB 4
#define NN 128
#define DD 128
#define HH 8
#define DKK 16
#define LL 3
#define EPSV 1e-5f

#define LDW 136   // ushort stride (272 B, 16B-multiple) for most bf16 matrices
#define LDK 160   // ushort stride (320 B) for Km with zero-padded cols 128..159
#define LDS_S 132 // float stride for score matrix

typedef __attribute__((ext_vector_type(16))) __bf16 v16bf;
typedef __attribute__((ext_vector_type(8))) float v8f;

union FragBF {
  v16bf v;
  uint4 q[2];
  unsigned short s[16];
};

__device__ __forceinline__ unsigned short f2bf(float f) {
  union { float f; unsigned int u; } x;
  x.f = f;
  unsigned int u = x.u;
  u += 0x7fffu + ((u >> 16) & 1u);  // round-to-nearest-even
  return (unsigned short)(u >> 16);
}

// A-matrix 16x32 bf16 fragment (ISA 7.12.2): lane half lh adds +8 within each
// 16-wide K half; per lane this is two contiguous 8-ushort (16B) chunks.
__device__ __forceinline__ v16bf loadFragA32(const unsigned short* A, int lda,
                                             int mBase, int kBase) {
  int lane = threadIdx.x & 31;
  const unsigned short* p =
      A + (mBase + (lane & 15)) * lda + kBase + ((lane >> 4) << 3);
  FragBF u;
  u.q[0] = *(const uint4*)p;
  u.q[1] = *(const uint4*)(p + 16);
  return u.v;
}

// K=16 variant: slots j>=4 (K>=16) are zero for every lane (uniform).
__device__ __forceinline__ v16bf loadFragA16(const unsigned short* A, int lda,
                                             int mBase, int kBase) {
  int lane = threadIdx.x & 31;
  const unsigned short* p =
      A + (mBase + (lane & 15)) * lda + kBase + ((lane >> 4) << 3);
  FragBF u;
  u.q[0] = *(const uint4*)p;
  u.q[1] = make_uint4(0u, 0u, 0u, 0u);
  return u.v;
}

// B-matrix 32x16 fragment from TRANSPOSED storage Bt[n][k]: lane (n, half)
// needs K=8*half..+7 and 16+8*half..+7 -> two 8-ushort (16B) chunks.
__device__ __forceinline__ v16bf loadFragBT(const unsigned short* Bt, int ldb,
                                            int nBase, int kBase) {
  int lane = threadIdx.x & 31;
  const unsigned short* p =
      Bt + (nBase + (lane & 15)) * ldb + kBase + ((lane >> 4) << 3);
  FragBF u;
  u.q[0] = *(const uint4*)p;
  u.q[1] = *(const uint4*)(p + 16);
  return u.v;
}

__device__ __forceinline__ v8f wmma_bf16(v16bf a, v16bf b, v8f c) {
  return __builtin_amdgcn_wmma_f32_16x16x32_bf16(false, a, false, b, (short)0, c,
                                                 false, false);
}

// ---------------- init embedding: e[b,n,m,d] = x_init[b,n,m,:] . W_init[d,:]
__global__ void __launch_bounds__(256)
init_embed_kernel(const float* __restrict__ x0, const float* __restrict__ x1,
                  const float* __restrict__ x2, const float* __restrict__ rf,
                  const float* __restrict__ Wi, float* __restrict__ e) {
  size_t idx = (size_t)blockIdx.x * 256 + threadIdx.x;
  size_t bnm = idx >> 7;
  int d = (int)(idx & 127);
  int b = (int)(idx >> 21);  // N*N*D = 2^21
  e[idx] = x0[bnm] * Wi[d * 4 + 0] + x1[bnm] * Wi[d * 4 + 1] +
           x2[bnm] * Wi[d * 4 + 2] + rf[b] * Wi[d * 4 + 3];
}

__global__ void zero_bn_kernel(float* bnS, float* bnQ) {
  int t = threadIdx.x;
  if (t < DD) bnS[t] = 0.f;
  else if (t < 2 * DD) bnQ[t - DD] = 0.f;
}

// ---------------- fused attention + FF sublayer for one layer.
// One block per (b, i). 256 threads = 8 waves; wave w owns output rows [16w,16w+16).
__global__ void __launch_bounds__(256)
attn_ff_kernel(const float* __restrict__ e, float* __restrict__ y,
               const float* __restrict__ Wq, const float* __restrict__ Wk,
               const float* __restrict__ Wv, const float* __restrict__ Wo,
               const float* __restrict__ Wff, const float* __restrict__ bff,
               float* __restrict__ bnSum, float* __restrict__ bnSqr) {
  extern __shared__ char smem[];
  unsigned short* sEq = (unsigned short*)(smem + 0);        // 34816 B
  unsigned short* sEk = (unsigned short*)(smem + 34816);    // 34816 B
  float* sS = (float*)(smem + 0);                           // aliases sEq/sEk
  unsigned short* sW = (unsigned short*)(smem + 69632);     // weights, TRANSPOSED [col][k]
  unsigned short* sQ = (unsigned short*)(smem + 104448);    // Qm row-major; reused as sM
  unsigned short* sKm = (unsigned short*)(smem + 139264);   // Km row-major [m][k], ld 160
  unsigned short* sVT = (unsigned short*)(smem + 180224);   // V transposed [kd][m]
  unsigned short* sP = (unsigned short*)(smem + 215040);    // exp(scores) row-major
  unsigned short* sC = (unsigned short*)(smem + 249856);    // ctx row-major
  float* red = (float*)(smem + 284672);   // 256 floats (reductions + bias)
  float* rsum = (float*)(smem + 285696);  // 128 floats
  float* psum = (float*)(smem + 286208);  // 128 floats
  float* psqr = (float*)(smem + 286720);  // 128 floats

  const int tid = threadIdx.x;
  const int w = tid >> 5;
  const int lane = tid & 31;
  const int lh = lane >> 4;
  const int ln = lane & 15;
  const int b = blockIdx.x >> 7;
  const int i = blockIdx.x & 127;

  if (tid < DD) { psum[tid] = 0.f; psqr[tid] = 0.f; }
  // zero the K-padding columns of sKm (cols 128..159)
  for (int t = tid; t < NN * 32; t += 256) {
    int m = t >> 5, c = 128 + (t & 31);
    sKm[m * LDK + c] = 0;
  }

  // Stage activations: K/V source row e[b,i,:,:]; Q source column e[b,:,i,:]
  const size_t eRow = ((size_t)(b * NN + i)) * NN * DD;
  for (int t = tid; t < NN * DD; t += 256) {
    int n = t >> 7, d = t & 127;
    sEk[n * LDW + d] = f2bf(e[eRow + (size_t)n * DD + d]);
    sEq[n * LDW + d] = f2bf(e[((size_t)(b * NN + n) * NN + i) * DD + d]);
  }
  // WqCat transposed: sW[c][d], c = h*16+k
  for (int t = tid; t < NN * DD; t += 256) {
    int dr = t >> 7, c = t & 127, h = c >> 4, k = c & 15;
    sW[c * LDW + dr] = f2bf(Wq[((size_t)h * DD + dr) * DKK + k]);
  }
  __syncthreads();

  // Qm = sEq @ WqCat -> sQ (bf16, N x 128, row-major)
  for (int nt = 0; nt < 8; ++nt) {
    v8f acc = {0.f, 0.f, 0.f, 0.f, 0.f, 0.f, 0.f, 0.f};
#pragma unroll
    for (int kt = 0; kt < 4; ++kt)
      acc = wmma_bf16(loadFragA32(sEq, LDW, 16 * w, 32 * kt),
                      loadFragBT(sW, LDW, 16 * nt, 32 * kt), acc);
#pragma unroll
    for (int r = 0; r < 8; ++r)
      sQ[(16 * w + r + 8 * lh) * LDW + 16 * nt + ln] = f2bf(acc[r]);
  }
  __syncthreads();

  // Km row-major [m][k] (this IS the transposed-B layout for Q.K^T)
  for (int t = tid; t < NN * DD; t += 256) {
    int dr = t >> 7, c = t & 127, h = c >> 4, k = c & 15;
    sW[c * LDW + dr] = f2bf(Wk[((size_t)h * DD + dr) * DKK + k]);
  }
  __syncthreads();
  for (int nt = 0; nt < 8; ++nt) {
    v8f acc = {0.f, 0.f, 0.f, 0.f, 0.f, 0.f, 0.f, 0.f};
#pragma unroll
    for (int kt = 0; kt < 4; ++kt)
      acc = wmma_bf16(loadFragA32(sEk, LDW, 16 * w, 32 * kt),
                      loadFragBT(sW, LDW, 16 * nt, 32 * kt), acc);
#pragma unroll
    for (int r = 0; r < 8; ++r)
      sKm[(16 * w + r + 8 * lh) * LDK + 16 * nt + ln] = f2bf(acc[r]);
  }
  __syncthreads();

  // Vm, stored TRANSPOSED: sVT[kd][m]
  for (int t = tid; t < NN * DD; t += 256) {
    int dr = t >> 7, c = t & 127, h = c >> 4, k = c & 15;
    sW[c * LDW + dr] = f2bf(Wv[((size_t)h * DD + dr) * DKK + k]);
  }
  __syncthreads();
  for (int nt = 0; nt < 8; ++nt) {
    v8f acc = {0.f, 0.f, 0.f, 0.f, 0.f, 0.f, 0.f, 0.f};
#pragma unroll
    for (int kt = 0; kt < 4; ++kt)
      acc = wmma_bf16(loadFragA32(sEk, LDW, 16 * w, 32 * kt),
                      loadFragBT(sW, LDW, 16 * nt, 32 * kt), acc);
#pragma unroll
    for (int r = 0; r < 8; ++r)
      sVT[(16 * nt + ln) * LDW + 16 * w + r + 8 * lh] = f2bf(acc[r]);
  }
  __syncthreads();
  // sEq/sEk now dead; sS aliases that region.

  // -------- per-head attention --------
  for (int h = 0; h < HH; ++h) {
    // scores S = Q_h @ K_h^T : single K=16 WMMA per tile.
    // A: sQ cols [16h,16h+16) (upper K zeroed uniformly);
    // B: sKm rows (transposed layout), reads run into zeroed pad cols.
    for (int nt = 0; nt < 8; ++nt) {
      v8f acc = {0.f, 0.f, 0.f, 0.f, 0.f, 0.f, 0.f, 0.f};
      acc = wmma_bf16(loadFragA16(sQ, LDW, 16 * w, 16 * h),
                      loadFragBT(sKm, LDK, 16 * nt, 16 * h), acc);
#pragma unroll
      for (int r = 0; r < 8; ++r)
        sS[(16 * w + r + 8 * lh) * LDS_S + 16 * nt + ln] = acc[r];
    }
    __syncthreads();

    // softmax rows (scale=1/sqrt(16)=0.25); store unnormalized exp in sP,
    // fold 1/rowsum into the ctx store.
    {
      int r = tid >> 1, hf = tid & 1;
      float lmax = -3.4e38f;
      for (int c = hf * 64; c < hf * 64 + 64; ++c)
        lmax = fmaxf(lmax, sS[r * LDS_S + c] * 0.25f);
      red[tid] = lmax;
      __syncthreads();
      float rmax = fmaxf(red[2 * r], red[2 * r + 1]);
      __syncthreads();
      float lsum = 0.f;
      for (int c = hf * 64; c < hf * 64 + 64; ++c) {
        float ev = __expf(sS[r * LDS_S + c] * 0.25f - rmax);
        sP[r * LDW + c] = f2bf(ev);
        lsum += ev;
      }
      red[tid] = lsum;
      __syncthreads();
      if (hf == 0) rsum[r] = red[2 * r] + red[2 * r + 1];
    }
    __syncthreads();

    // ctx_h = P @ V[:, 16h:16h+16] / rowsum -> sC[:, 16h:16h+16]
    {
      v8f acc = {0.f, 0.f, 0.f, 0.f, 0.f, 0.f, 0.f, 0.f};
#pragma unroll
      for (int kt = 0; kt < 4; ++kt)
        acc = wmma_bf16(loadFragA32(sP, LDW, 16 * w, 32 * kt),
                        loadFragBT(sVT, LDW, 16 * h, 32 * kt), acc);
#pragma unroll
      for (int r = 0; r < 8; ++r) {
        int m = 16 * w + r + 8 * lh;
        sC[m * LDW + 16 * h + ln] = f2bf(acc[r] / rsum[m]);
      }
    }
    __syncthreads();
  }

  // -------- output projection: mha = sC @ WoutCat -> sM (reuse sQ) --------
  // Wout[l] flat is [H*DK, D]; transposed into sW[d][r].
  for (int t = tid; t < NN * DD; t += 256) {
    int rr = t >> 7, c = t & 127;
    sW[c * LDW + rr] = f2bf(Wo[t]);
  }
  __syncthreads();
  unsigned short* sM = sQ;
  for (int nt = 0; nt < 8; ++nt) {
    v8f acc = {0.f, 0.f, 0.f, 0.f, 0.f, 0.f, 0.f, 0.f};
#pragma unroll
    for (int kt = 0; kt < 4; ++kt)
      acc = wmma_bf16(loadFragA32(sC, LDW, 16 * w, 32 * kt),
                      loadFragBT(sW, LDW, 16 * nt, 32 * kt), acc);
#pragma unroll
    for (int r = 0; r < 8; ++r)
      sM[(16 * w + r + 8 * lh) * LDW + 16 * nt + ln] = f2bf(acc[r]);
  }
  __syncthreads();

  // -------- FF sublayer: H = mha @ Wff^T + bff; y = relu(H) + e --------
  // Transposed-B of Wff^T is Wff itself: straight copy with stride remap.
  for (int t = tid; t < NN * DD; t += 256) {
    int ep = t >> 7, d = t & 127;
    sW[ep * LDW + d] = f2bf(Wff[t]);
  }
  if (tid < DD) red[128 + tid] = bff[tid];
  __syncthreads();
  for (int nt = 0; nt < 8; ++nt) {
    v8f acc = {0.f, 0.f, 0.f, 0.f, 0.f, 0.f, 0.f, 0.f};
#pragma unroll
    for (int kt = 0; kt < 4; ++kt)
      acc = wmma_bf16(loadFragA32(sM, LDW, 16 * w, 32 * kt),
                      loadFragBT(sW, LDW, 16 * nt, 32 * kt), acc);
    int ep = 16 * nt + ln;
    float bias = red[128 + ep];
#pragma unroll
    for (int r = 0; r < 8; ++r) {
      int n = 16 * w + r + 8 * lh;
      float v = acc[r] + bias;
      v = v > 0.f ? v : 0.f;
      size_t g = ((size_t)(b * NN + n) * NN + i) * DD + ep;
      v += e[g];  // residual (layer input, not yet overwritten)
      y[g] = v;
      atomicAdd(&psum[ep], v);
      atomicAdd(&psqr[ep], v * v);
    }
  }
  __syncthreads();
  if (tid < DD) atomicAdd(&bnSum[tid], psum[tid]);
  else if (tid < 2 * DD) atomicAdd(&bnSqr[tid - DD], psqr[tid - DD]);
}

// ---------------- BatchNorm finalize + apply (training-mode batch stats)
__global__ void __launch_bounds__(256)
bn_apply_kernel(const float* __restrict__ y, const float* __restrict__ bnSum,
                const float* __restrict__ bnSqr, const float* __restrict__ gamma,
                const float* __restrict__ beta, float* __restrict__ e) {
  size_t idx = (size_t)blockIdx.x * 256 + threadIdx.x;
  int d = (int)(idx & 127);
  const float inv = 1.0f / (float)(BB * NN * NN);
  float mu = bnSum[d] * inv;
  float var = bnSqr[d] * inv - mu * mu;
  float s = rsqrtf(var + EPSV) * gamma[d];
  e[idx] = (y[idx] - mu) * s + beta[d];
}

// ---------------- outputs
__global__ void node_mean_kernel(const float* __restrict__ e, float* __restrict__ out) {
  int bn = blockIdx.x;  // b*N + n
  int d = threadIdx.x;
  float s = 0.f;
  for (int m = 0; m < NN; ++m)
    s += e[((size_t)bn * NN + m) * DD + d];
  out[(size_t)bn * DD + d] = s * (1.0f / NN);
}

__global__ void pool_mean_kernel(const float* __restrict__ nodeEmb, float* __restrict__ out) {
  int b = blockIdx.x;
  int d = threadIdx.x;
  float s = 0.f;
  for (int n = 0; n < NN; ++n)
    s += nodeEmb[((size_t)b * NN + n) * DD + d];
  out[(size_t)b * DD + d] = s * (1.0f / NN);
}

extern "C" void kernel_launch(void* const* d_in, const int* in_sizes, int n_in,
                              void* d_out, int out_size, void* d_ws, size_t ws_size,
                              hipStream_t stream) {
  const float* x0 = (const float*)d_in[0];
  const float* x1 = (const float*)d_in[1];
  const float* x2 = (const float*)d_in[2];
  const float* rf = (const float*)d_in[3];
  const float* Wi = (const float*)d_in[4];
  const float* Wq = (const float*)d_in[5];
  const float* Wk = (const float*)d_in[6];
  const float* Wv = (const float*)d_in[7];
  const float* Wo = (const float*)d_in[8];
  const float* Wff = (const float*)d_in[9];
  const float* bff = (const float*)d_in[10];
  const float* gamma = (const float*)d_in[11];
  const float* beta = (const float*)d_in[12];

  const size_t EL = (size_t)BB * NN * NN * DD;  // 8388608
  float* e = (float*)d_ws;
  float* y = e + EL;
  float* bnS = y + EL;
  float* bnQ = bnS + DD;

  const unsigned SMEM = 287232;  // dynamic LDS bytes (< 320KB WGP limit)
  const int nElemBlocks = (int)(EL / 256);  // 32768

  init_embed_kernel<<<nElemBlocks, 256, 0, stream>>>(x0, x1, x2, rf, Wi, e);

  const size_t WSTR = (size_t)HH * DD * DKK;  // 16384 per layer, all weights
  for (int l = 0; l < LL; ++l) {
    zero_bn_kernel<<<1, 256, 0, stream>>>(bnS, bnQ);
    attn_ff_kernel<<<BB * NN, 256, SMEM, stream>>>(
        e, y, Wq + (size_t)l * WSTR, Wk + (size_t)l * WSTR, Wv + (size_t)l * WSTR,
        Wo + (size_t)l * WSTR, Wff + (size_t)l * DD * DD, bff + (size_t)l * DD,
        bnS, bnQ);
    bn_apply_kernel<<<nElemBlocks, 256, 0, stream>>>(y, bnS, bnQ,
                                                     gamma + (size_t)l * DD,
                                                     beta + (size_t)l * DD, e);
  }

  float* out = (float*)d_out;
  node_mean_kernel<<<BB * NN, 128, 0, stream>>>(e, out);
  pool_mean_kernel<<<BB, 128, 0, stream>>>(out, out + (size_t)BB * NN * DD);
}
